// deepirt_dropout_77154792505410
// MI455X (gfx1250) — compile-verified, matrix operand order checked by reading, not run on hardware
//
#include <hip/hip_runtime.h>
#include <hip/hip_bf16.h>
#include <math.h>

typedef __attribute__((ext_vector_type(16))) _Float16 v16h;
typedef __attribute__((ext_vector_type(8)))  float    v8f;

#define BB 64
#define TT 256
#define BT (BB * TT)     // 16384
#define MM 64
#define DKK 128
#define DVV 256
#define SSS 128
#define NSK 200
#define NQ  2000

__device__ __forceinline__ int laneid() { return (int)(threadIdx.x & 31); }

// Fragment for v_wmma_f32_16x16x32_f16 from a row-major f16 row with
// contiguous K. ISA 16-bit A 16x32 layout:
//   lane<16 : K = {0..7, 16..23},  lane>=16 : K = {8..15, 24..31}
// (each 8-element chunk is one 16-byte load)
__device__ __forceinline__ v16h frag_contig(const _Float16* rowp, int k0) {
  int kb = k0 + ((laneid() & 16) ? 8 : 0);
  v16h f;
#pragma unroll
  for (int i = 0; i < 8; ++i) f[i] = rowp[kb + i];
#pragma unroll
  for (int i = 0; i < 8; ++i) f[i + 8] = rowp[kb + 16 + i];
  return f;
}

// ---------- one-time operand prep (f32 -> f16, and K x N -> N x K f16) ----------

__global__ void cvt_f16_kernel(const float* __restrict__ src,
                               _Float16* __restrict__ dst, int n) {
  int i = (int)(blockIdx.x * 256 + threadIdx.x);
  if (i < n) dst[i] = (_Float16)src[i];
}

// Wt[n*K + k] = W[k*N + n]
__global__ void transpose_f16_kernel(const float* __restrict__ W,
                                     _Float16* __restrict__ Wt, int K, int N) {
  int i = (int)(blockIdx.x * 256 + threadIdx.x);
  if (i < K * N) {
    int n = i / K, k = i - n * K;
    Wt[i] = (_Float16)W[(size_t)k * N + n];
  }
}

// ---------- corr = softmax(Eq[q] @ Kmem^T).  Block: 32 rows x 64 cols ----------

__global__ void corr_kernel(const int* __restrict__ qidx,
                            const _Float16* __restrict__ Eq16,
                            const _Float16* __restrict__ Kmem16,
                            float* __restrict__ corr) {
  __shared__ float logits[32][MM];
  int wave = (int)(threadIdx.x >> 5), lane = laneid();
  int rowTile = wave >> 2, colTile = wave & 3;
  int r15 = lane & 15;
  int arow = (int)blockIdx.x * 32 + rowTile * 16 + r15;
  const _Float16* aptr = Eq16 + (size_t)qidx[arow] * DKK;
  const _Float16* bptr = Kmem16 + (size_t)(colTile * 16 + r15) * DKK;
  v8f acc = {};
#pragma unroll
  for (int ks = 0; ks < DKK / 32; ++ks) {
    v16h a = frag_contig(aptr, ks * 32);
    v16h b = frag_contig(bptr, ks * 32);
    acc = __builtin_amdgcn_wmma_f32_16x16x32_f16(false, a, false, b, (short)0, acc,
                                                 false, false);
  }
#pragma unroll
  for (int r = 0; r < 8; ++r) {
    int row = rowTile * 16 + r + ((lane & 16) ? 8 : 0);
    logits[row][colTile * 16 + r15] = acc[r];
  }
  __syncthreads();
  if (threadIdx.x < 32) {
    int row = (int)threadIdx.x;
    float mx = -3.4e38f;
    for (int m = 0; m < MM; ++m) mx = fmaxf(mx, logits[row][m]);
    float s = 0.f;
    for (int m = 0; m < MM; ++m) s += expf(logits[row][m] - mx);
    float inv = 1.f / s;
    size_t base = (size_t)((int)blockIdx.x * 32 + row) * MM;
    for (int m = 0; m < MM; ++m) corr[base + m] = expf(logits[row][m] - mx) * inv;
  }
}

// ---------- out = Etab[idx] @ W + bias (Wt is N x K f16) ----------

__global__ void gemm_embed_kernel(const int* __restrict__ idx,
                                  const _Float16* __restrict__ Etab16, int ek,
                                  const _Float16* __restrict__ Wt,
                                  const float* __restrict__ bias, int N,
                                  float* __restrict__ out) {
  int wave = (int)(threadIdx.x >> 5), lane = laneid();
  int rowTile = wave >> 2, colTile = wave & 3;
  int r15 = lane & 15;
  int arow = (int)blockIdx.x * 32 + rowTile * 16 + r15;
  const _Float16* aptr = Etab16 + (size_t)idx[arow] * ek;
  int ng = (int)blockIdx.y * 64 + colTile * 16 + r15;
  const _Float16* bptr = Wt + (size_t)ng * ek;
  v8f acc = {};
  for (int ks = 0; ks < ek / 32; ++ks) {
    v16h a = frag_contig(aptr, ks * 32);
    v16h b = frag_contig(bptr, ks * 32);
    acc = __builtin_amdgcn_wmma_f32_16x16x32_f16(false, a, false, b, (short)0, acc,
                                                 false, false);
  }
  float bv = bias[ng];
#pragma unroll
  for (int r = 0; r < 8; ++r) {
    int row = (int)blockIdx.x * 32 + rowTile * 16 + r + ((lane & 16) ? 8 : 0);
    out[(size_t)row * N + ng] = acc[r] + bv;
  }
}

// ---------- out[row] = tanh(Etab[idx] @ W1 + b1) @ w2 + b2 ----------
// Block: 16 rows x 128 cols (S), 8 waves.  W1t is S x K f16.

__global__ void mlp_scalar_kernel(const int* __restrict__ idx,
                                  const _Float16* __restrict__ Etab16,
                                  const _Float16* __restrict__ W1t,
                                  const float* __restrict__ b1,
                                  const float* __restrict__ w2,
                                  const float* __restrict__ b2,
                                  float* __restrict__ out) {
  __shared__ float part[16];
  int wave = (int)(threadIdx.x >> 5), lane = laneid();
  int r15 = lane & 15;
  int arow = (int)blockIdx.x * 16 + r15;
  const _Float16* aptr = Etab16 + (size_t)idx[arow] * DKK;
  int ng = wave * 16 + r15;
  const _Float16* bptr = W1t + (size_t)ng * DKK;
  v8f acc = {};
#pragma unroll
  for (int ks = 0; ks < DKK / 32; ++ks) {
    v16h a = frag_contig(aptr, ks * 32);
    v16h b = frag_contig(bptr, ks * 32);
    acc = __builtin_amdgcn_wmma_f32_16x16x32_f16(false, a, false, b, (short)0, acc,
                                                 false, false);
  }
  if (threadIdx.x < 16) part[threadIdx.x] = 0.f;
  __syncthreads();
  float bv = b1[ng], wv = w2[ng];
#pragma unroll
  for (int r = 0; r < 8; ++r) {
    int row = r + ((lane & 16) ? 8 : 0);
    atomicAdd(&part[row], tanhf(acc[r] + bv) * wv);
  }
  __syncthreads();
  if (threadIdx.x < 16)
    out[(size_t)blockIdx.x * 16 + threadIdx.x] = part[threadIdx.x] + b2[0];
}

// ---------- sequential scan over T; Mv (64x256 f32) resident in LDS ----------

__global__ void scan_kernel(const float* __restrict__ corr,
                            const float* __restrict__ erase,
                            const float* __restrict__ add,
                            const float* __restrict__ Vmem,
                            const float* __restrict__ Wsa,
                            const float* __restrict__ bsa,
                            const float* __restrict__ qd,
                            const float* __restrict__ sd,
                            float* __restrict__ out) {
  extern __shared__ float smem[];
  float* Mv  = smem;            // MM*DVV
  float* cw  = Mv + MM * DVV;   // MM
  float* er  = cw + MM;         // DVV
  float* ad  = er + DVV;        // DVV
  float* rd  = ad + DVV;        // DVV
  float* red = rd + DVV;        // 8 wave partials
  int b = (int)blockIdx.x;
  int tid = (int)threadIdx.x;
  int lane = laneid(), wave = tid >> 5;
  for (int i = tid; i < MM * DVV; i += 256) Mv[i] = Vmem[i];
  __syncthreads();
  for (int t = 0; t < TT; ++t) {
    size_t bt = (size_t)b * TT + t;
    if (tid < MM) cw[tid] = corr[bt * MM + tid];
    er[tid] = erase[bt * DVV + tid];
    ad[tid] = add[bt * DVV + tid];
    __syncthreads();
    // read[d] = sum_m cw[m] * Mv[m][d]  (thread d owns column d)
    float acc = 0.f;
#pragma unroll 4
    for (int m = 0; m < MM; ++m) acc += cw[m] * Mv[m * DVV + tid];
    rd[tid] = acc;
    // update column d (each thread touches only its own column -> no hazard)
    float e = er[tid], a = ad[tid];
#pragma unroll 4
    for (int m = 0; m < MM; ++m) {
      float c = cw[m];
      float old = Mv[m * DVV + tid];
      Mv[m * DVV + tid] = old * (1.f - e * c) + a * c;
    }
    // sa = sum_d rd[d] * (sum_m cw[m]*Wsa[d][m]) + sum_m cw[m]*bsa[m]
    float u = 0.f;
    const float* wrow = Wsa + (size_t)tid * MM;
#pragma unroll 4
    for (int m = 0; m < MM; ++m) u += cw[m] * wrow[m];
    float p = rd[tid] * u;
    if (tid < MM) p += cw[tid] * bsa[tid];
    // wave32 shuffle reduction, then cross-wave via LDS
#pragma unroll
    for (int off = 16; off > 0; off >>= 1) p += __shfl_down(p, off, 32);
    if (lane == 0) red[wave] = p;
    __syncthreads();
    if (tid == 0) {
      float sa = red[0];
#pragma unroll
      for (int w = 1; w < 8; ++w) sa += red[w];
      float z = 3.f * sa - qd[bt] - sd[bt];
      out[bt] = 1.f / (1.f + expf(-z));
    }
    __syncthreads();
  }
}

extern "C" void kernel_launch(void* const* d_in, const int* in_sizes, int n_in,
                              void* d_out, int out_size, void* d_ws, size_t ws_size,
                              hipStream_t stream) {
  const int*   q_data  = (const int*)d_in[0];
  const int*   qa_data = (const int*)d_in[1];
  const int*   s_data  = (const int*)d_in[2];
  const float* Kmem    = (const float*)d_in[3];
  const float* Vmem    = (const float*)d_in[4];
  const float* Eq      = (const float*)d_in[5];
  const float* Eqa     = (const float*)d_in[6];
  const float* Es      = (const float*)d_in[7];
  const float* We      = (const float*)d_in[8];
  const float* be      = (const float*)d_in[9];
  const float* Wa      = (const float*)d_in[10];
  const float* ba      = (const float*)d_in[11];
  const float* Wsa     = (const float*)d_in[12];
  const float* bsa     = (const float*)d_in[13];
  const float* Wqd1    = (const float*)d_in[14];
  const float* bqd1    = (const float*)d_in[15];
  const float* Wqd     = (const float*)d_in[16];
  const float* bqd     = (const float*)d_in[17];
  const float* Wsd1    = (const float*)d_in[18];
  const float* bsd1    = (const float*)d_in[19];
  const float* Wsd     = (const float*)d_in[20];
  const float* bsd     = (const float*)d_in[21];

  // ---- workspace layout: f32 intermediates, then f16 operand copies ----
  float* ws    = (float*)d_ws;
  float* corr  = ws;  ws += (size_t)BT * MM;
  float* erase = ws;  ws += (size_t)BT * DVV;
  float* add   = ws;  ws += (size_t)BT * DVV;
  float* qd    = ws;  ws += BT;
  float* sd    = ws;  ws += BT;

  _Float16* hp = (_Float16*)ws;
  const int nEq   = (NSK + 1) * DKK;       // 201*128
  const int nEqa  = (2 * NSK + 1) * DVV;   // 401*256
  const int nEs   = (NQ + 1) * DKK;        // 2001*128
  const int nKm   = MM * DKK;              // 64*128
  const int nWe   = DVV * DVV;             // 256*256
  const int nW1   = DKK * SSS;             // 128*128
  _Float16* Eq16   = hp;  hp += nEq;
  _Float16* Eqa16  = hp;  hp += nEqa;
  _Float16* Es16   = hp;  hp += nEs;
  _Float16* Kmem16 = hp;  hp += nKm;
  _Float16* WeT    = hp;  hp += nWe;
  _Float16* WaT    = hp;  hp += nWe;
  _Float16* Wqd1T  = hp;  hp += nW1;
  _Float16* Wsd1T  = hp;  hp += nW1;
  float* out = (float*)d_out;

  // operand prep (tiny, all L2-resident)
  cvt_f16_kernel<<<(nEq  + 255) / 256, 256, 0, stream>>>(Eq,   Eq16,   nEq);
  cvt_f16_kernel<<<(nEqa + 255) / 256, 256, 0, stream>>>(Eqa,  Eqa16,  nEqa);
  cvt_f16_kernel<<<(nEs  + 255) / 256, 256, 0, stream>>>(Es,   Es16,   nEs);
  cvt_f16_kernel<<<(nKm  + 255) / 256, 256, 0, stream>>>(Kmem, Kmem16, nKm);
  transpose_f16_kernel<<<(nWe + 255) / 256, 256, 0, stream>>>(We,   WeT,   DVV, DVV);
  transpose_f16_kernel<<<(nWe + 255) / 256, 256, 0, stream>>>(Wa,   WaT,   DVV, DVV);
  transpose_f16_kernel<<<(nW1 + 255) / 256, 256, 0, stream>>>(Wqd1, Wqd1T, DKK, SSS);
  transpose_f16_kernel<<<(nW1 + 255) / 256, 256, 0, stream>>>(Wsd1, Wsd1T, DKK, SSS);

  // WMMA GEMM phase
  corr_kernel<<<BT / 32, 256, 0, stream>>>(q_data, Eq16, Kmem16, corr);
  gemm_embed_kernel<<<dim3(BT / 32, DVV / 64), 256, 0, stream>>>(
      qa_data, Eqa16, DVV, WeT, be, DVV, erase);
  gemm_embed_kernel<<<dim3(BT / 32, DVV / 64), 256, 0, stream>>>(
      qa_data, Eqa16, DVV, WaT, ba, DVV, add);
  mlp_scalar_kernel<<<BT / 16, 256, 0, stream>>>(q_data, Eq16, Wqd1T, bqd1, Wqd, bqd, qd);
  mlp_scalar_kernel<<<BT / 16, 256, 0, stream>>>(s_data, Es16, Wsd1T, bsd1, Wsd, bsd, sd);

  // sequential scan (Mv in LDS, dynamic to avoid static-LDS cap)
  size_t smem = (size_t)(MM * DVV + MM + 4 * DVV) * sizeof(float);
  scan_kernel<<<BB, 256, smem, stream>>>(corr, erase, add, Vmem, Wsa, bsa, qd, sd, out);
}
